// HeatmapDepthModel_46943992546201
// MI455X (gfx1250) — matrix operand order repaired and checked by reading, not compiled, not run on hardware
//
#include <hip/hip_runtime.h>
#include <hip/hip_bf16.h>

// ---------------------------------------------------------------------------
// HeatmapDepthModel fused pipeline for gfx1250 (MI455X), wave32 + WMMA bf16.
// conv1 (implicit GEMM 65536x256xK, ~34 GFLOP) -> v_wmma_f32_16x16x32_bf16.
// K reordered as (kernel-pos, channel-padded-to-128) so all A-fragment LDS
// addresses are affine (no div/mod in the inner loop); weights pre-transposed
// to bf16 in that order by a one-shot kernel. conv2 (1x1) fused into the
// epilogue. Input patches staged to LDS with GLOBAL_LOAD_ASYNC_TO_LDS_B128
// (ASYNCcnt); weights streamed from L2 with global_prefetch_b8.
// ---------------------------------------------------------------------------

typedef __attribute__((ext_vector_type(16))) __bf16 v16bf;
typedef __attribute__((ext_vector_type(8)))  float  v8f;

#define C1     112
#define CPAD   128             // channels padded for affine K addressing
#define KT     1152            // 9 kernel positions * 128 channels
#define ROWSTR 40              // padded patch row stride (floats)
#define CSTR   160             // 4 rows * 40
#define PATCHF (CPAD*CSTR)     // 20480 floats = 80 KB (c 112..127 zeroed)
#define SIG2I  0.25f           // 1/sigma^2, sigma=2

static __device__ __forceinline__ int iclamp(int v, int lo, int hi) {
    return v < lo ? lo : (v > hi ? hi : v);
}

// ---------------------------------------------------------------------------
// Kernel 0: one-shot weight transpose+convert:
// w1t[n][kp][c] (bf16, c padded to 128) = conv1_w[n][c][ky][kx] (f32).
// ---------------------------------------------------------------------------
__global__ void wtrans_kernel(const float* __restrict__ w1,
                              __bf16* __restrict__ w1t) {
    int i = blockIdx.x * blockDim.x + threadIdx.x;      // over 256*1152
    if (i >= 256 * KT) return;
    int n  = i / KT;
    int r  = i - n * KT;
    int kp = r >> 7;
    int c  = r & 127;
    float v = (c < C1) ? w1[n * 1008 + c * 9 + kp] : 0.f;
    w1t[i] = (__bf16)v;
}

// ---------------------------------------------------------------------------
// Kernel 1: conv1(3x3,112->256)+bias+ReLU fused with conv2(1x1,256->1)+bias.
// Grid: 64 images * 16 row-pairs. Block: 128 threads (4 waves).
// Wave w owns channels [w*64, w*64+64) (4 N-subtiles) for all 4 M-tiles.
// ---------------------------------------------------------------------------
__global__ __launch_bounds__(128) void conv_fused_kernel(
    const float* __restrict__ x,      // (64,112,32,32)
    const __bf16* __restrict__ w1t,   // (256,9,128) bf16
    const float* __restrict__ b1,     // (256)
    const float* __restrict__ w2,     // (256)
    const float* __restrict__ b2,     // (1)
    float* __restrict__ logits)       // (64,32,32)
{
    __shared__ float s_patchf[PATCHF];   // [c][dy 0..3][dx 0..39], interior dx 4..35
    __shared__ float s_logit[64];

    const int b    = blockIdx.x >> 4;
    const int y0   = (blockIdx.x & 15) << 1;          // rows y0, y0+1
    const int tid  = threadIdx.x;
    const int lane = tid & 31;
    const int wave = tid >> 5;
    const int half = lane >> 4;
    const int lm   = lane & 15;

    // --- zero guard columns (dx=3 / dx=36), pad channels, invalid edge rows ---
    for (int i = tid; i < C1 * 4 * 2; i += 128) {
        int c = i >> 3, r = i & 7;
        int dy = r >> 1, dx = (r & 1) ? 36 : 3;
        s_patchf[c * CSTR + dy * ROWSTR + dx] = 0.f;
    }
    for (int i = tid; i < (CPAD - C1) * CSTR; i += 128)
        s_patchf[C1 * CSTR + i] = 0.f;                 // channels 112..127
    if (y0 == 0) {
        for (int i = tid; i < C1 * 34; i += 128) {
            int c = i / 34, dx = i - c * 34;
            s_patchf[c * CSTR + 0 * ROWSTR + 3 + dx] = 0.f;
        }
    }
    if (y0 == 30) {
        for (int i = tid; i < C1 * 34; i += 128) {
            int c = i / 34, dx = i - c * 34;
            s_patchf[c * CSTR + 3 * ROWSTR + 3 + dx] = 0.f;
        }
    }
    if (tid < 64) s_logit[tid] = 0.f;

    // --- async-stage valid interior rows: memory -> LDS, 16B granules ---
    {
        const int dylo  = (y0 == 0)  ? 1 : 0;
        const int dyhi  = (y0 == 30) ? 2 : 3;
        const int nrows = dyhi - dylo + 1;
        const int totalv = C1 * nrows * 8;             // float4 units
        const uint32_t lds_base = (uint32_t)(uintptr_t)&s_patchf[0];
        for (int i = tid; i < totalv; i += 128) {
            int c  = i / (nrows * 8);
            int r  = i - c * (nrows * 8);
            int dy = dylo + (r >> 3);
            int q  = r & 7;                            // float4 index in row
            int yy = y0 - 1 + dy;
            uint32_t goff  = (uint32_t)((((b * C1 + c) * 32 + yy) * 32 + q * 4) * 4);
            uint32_t laddr = lds_base + (uint32_t)((c * CSTR + dy * ROWSTR + 4 + q * 4) * 4);
            asm volatile("global_load_async_to_lds_b128 %0, %1, %2"
                         :: "v"(laddr), "v"(goff), "s"(x) : "memory");
        }
        asm volatile("s_wait_asynccnt 0x0" ::: "memory");
    }
    __syncthreads();

    // --- WMMA main loop: 9 kernel positions x 4 channel-chunks of 32 ---
    v8f acc[4][4] = {};                    // [m_tile][n_subtile], 128 VGPRs
    const int nbase = wave * 64;

    for (int kp = 0; kp < 9; ++kp) {
        const int ky = kp / 3;             // loop-invariant-ish, 9 iterations
        const int kx = kp - ky * 3;
        for (int cc = 0; cc < 4; ++cc) {
            const int c0 = cc * 32;

            // Four A fragments (16x32 bf16), affine LDS addressing:
            // lane element j maps to channel c0 + half*16 + j.
            v16bf afr[4];
#pragma unroll
            for (int mt = 0; mt < 4; ++mt) {
                const int row = mt >> 1;
                const int col = (mt & 1) * 16 + lm;
                const float* ap = s_patchf + (c0 + half * 16) * CSTR
                                + (row + ky) * ROWSTR + col + kx + 3;
#pragma unroll
                for (int j = 0; j < 16; ++j) {
                    const int co = j;     // compile-time
                    afr[mt][j] = (__bf16)ap[co * CSTR];
                }
            }

#pragma unroll
            for (int s = 0; s < 4; ++s) {
                const int n = nbase + s * 16 + lm;
                const __bf16* wp = w1t + (size_t)n * KT + kp * CPAD + c0 + half * 16;
                __builtin_prefetch(wp + 32, 0, 1);     // -> global_prefetch_b8
                v16bf bf;                               // B: lane col = lm
#pragma unroll
                for (int j = 0; j < 16; ++j) bf[j] = wp[j];
#pragma unroll
                for (int mt = 0; mt < 4; ++mt)
                    acc[mt][s] = __builtin_amdgcn_wmma_f32_16x16x32_bf16(
                        false, afr[mt], false, bf, (short)0, acc[mt][s], false, false);
            }
        }
    }

    // --- epilogue: bias + ReLU + 1x1 conv dot, reduce over channels ---
#pragma unroll
    for (int mt = 0; mt < 4; ++mt) {
        float pr[8];
#pragma unroll
        for (int r = 0; r < 8; ++r) pr[r] = 0.f;
#pragma unroll
        for (int s = 0; s < 4; ++s) {
            int n = nbase + s * 16 + lm;
            float bias = b1[n];
            float wo   = w2[n];
#pragma unroll
            for (int r = 0; r < 8; ++r) {
                float v = acc[mt][s][r] + bias;
                v = v > 0.f ? v : 0.f;
                pr[r] += v * wo;
            }
        }
#pragma unroll
        for (int r = 0; r < 8; ++r) {
            float v = pr[r];
            v += __shfl_xor(v, 1, 32);
            v += __shfl_xor(v, 2, 32);
            v += __shfl_xor(v, 4, 32);
            v += __shfl_xor(v, 8, 32);
            pr[r] = v;
        }
        if (lm == 0) {
#pragma unroll
            for (int r = 0; r < 8; ++r)
                atomicAdd(&s_logit[mt * 16 + half * 8 + r], pr[r]);  // ds_add_f32
        }
    }
    __syncthreads();
    if (tid < 64) {
        int mt = tid >> 4, m = tid & 15;
        logits[(b * 32 + y0 + (mt >> 1)) * 32 + (mt & 1) * 16 + m] =
            s_logit[tid] + b2[0];
    }
}

// ---------------------------------------------------------------------------
// Kernel 2: spatial mean pool of z_feats (64,960,16,16) -> zp (64,960).
// ---------------------------------------------------------------------------
__global__ void zpool_kernel(const float* __restrict__ zf,
                             float* __restrict__ zp) {
    int idx = blockIdx.x * blockDim.x + threadIdx.x;
    if (idx >= 64 * 960) return;
    const float4* p = (const float4*)(zf + (size_t)idx * 256);
    float s = 0.f;
#pragma unroll 4
    for (int i = 0; i < 64; ++i) {
        float4 v = p[i];
        s += v.x + v.y + v.z + v.w;
    }
    zp[idx] = s * (1.f / 256.f);
}

// ---------------------------------------------------------------------------
// Kernel 3: fc1 (64x960 @ 960x256) + bias + ReLU, bf16 WMMA.
// ---------------------------------------------------------------------------
__global__ __launch_bounds__(32) void fc1_kernel(
    const float* __restrict__ zp,     // (64,960)
    const float* __restrict__ w,      // (256,960)
    const float* __restrict__ bias,   // (256)
    float* __restrict__ h)            // (64,256)
{
    const int m0   = (blockIdx.x >> 4) << 4;
    const int n0   = (blockIdx.x & 15) << 4;
    const int lane = threadIdx.x & 31;
    const int half = lane >> 4;
    const int lm   = lane & 15;

    v8f acc = {};
    for (int kc = 0; kc < 30; ++kc) {
        const int k0 = kc * 32;
        v16bf a, bf;
        const float* ap = zp + (size_t)(m0 + lm) * 960 + k0 + half * 16;
#pragma unroll
        for (int j = 0; j < 16; ++j) {
            const int co = j;
            a[j] = (__bf16)ap[co];
        }
        const float* wp = w + (size_t)(n0 + lm) * 960 + k0 + half * 16;
        __builtin_prefetch(wp + 32, 0, 1);
#pragma unroll
        for (int j = 0; j < 16; ++j) bf[j] = (__bf16)wp[j];
        acc = __builtin_amdgcn_wmma_f32_16x16x32_bf16(
            false, a, false, bf, (short)0, acc, false, false);
    }
    const int n  = n0 + lm;
    const float bb = bias[n];
#pragma unroll
    for (int r = 0; r < 8; ++r) {
        int m = m0 + half * 8 + r;
        float v = acc[r] + bb;
        h[m * 256 + n] = v > 0.f ? v : 0.f;
    }
}

// ---------------------------------------------------------------------------
// Kernel 4: fc2 (256->1) + z outputs + per-batch z Huber partial.
// ---------------------------------------------------------------------------
__global__ void fc2_kernel(const float* __restrict__ h,
                           const float* __restrict__ w,
                           const float* __restrict__ bias,
                           const float* __restrict__ zgt,
                           float* __restrict__ out,
                           float* __restrict__ ws_z) {
    int b = blockIdx.x * blockDim.x + threadIdx.x;
    if (b >= 64) return;
    float s = bias[0];
#pragma unroll 4
    for (int n = 0; n < 256; ++n) s += h[b * 256 + n] * w[n];
    out[256 + b] = s;                       // z_microns (Z_STD=1, Z_MEAN=0)
    float d = fabsf(s - zgt[b]);
    ws_z[b] = (d < 1.f) ? 0.5f * d * d : d - 0.5f;
}

// ---------------------------------------------------------------------------
// Kernel 5: per-batch heatmap post-processing.
// ---------------------------------------------------------------------------
static __device__ __forceinline__ float bilin(const float* s_log, int ix, int iy) {
    float u = (ix + 0.5f) * 0.25f - 0.5f;
    float v = (iy + 0.5f) * 0.25f - 0.5f;
    int xlo = (int)floorf(u); float fx = u - (float)xlo;
    int ylo = (int)floorf(v); float fy = v - (float)ylo;
    int x0c = iclamp(xlo, 0, 31), x1c = iclamp(xlo + 1, 0, 31);
    int y0c = iclamp(ylo, 0, 31), y1c = iclamp(ylo + 1, 0, 31);
    float a = s_log[y0c * 32 + x0c], bq = s_log[y0c * 32 + x1c];
    float c = s_log[y1c * 32 + x0c], d  = s_log[y1c * 32 + x1c];
    return a * (1.f - fx) * (1.f - fy) + bq * fx * (1.f - fy)
         + c * (1.f - fx) * fy + d * fx * fy;
}

__global__ __launch_bounds__(256) void heatmap_kernel(
    const float* __restrict__ logits,     // (64,32,32)
    const float* __restrict__ xy,         // (64,2)
    float* __restrict__ out,
    float* __restrict__ ws_fl,
    float* __restrict__ ws_np,
    float* __restrict__ ws_xy) {
    __shared__ float s_log[1024];
    __shared__ float s_red[256];
    __shared__ int   s_redi[256];
    const int b = blockIdx.x, tid = threadIdx.x;

    for (int i = tid; i < 1024; i += 256) s_log[i] = logits[b * 1024 + i];
    __syncthreads();

    // Pass 1: max logit (softmax stabilizer; argmax of sigmoid == argmax of logit).
    float mval = -1e30f; int midx = 0;
    for (int i = tid; i < 16384; i += 256) {
        int ix = i & 127, iy = i >> 7;
        float l = bilin(s_log, ix, iy);
        if (l > mval) { mval = l; midx = i; }
    }
    s_red[tid] = mval; s_redi[tid] = midx;
    __syncthreads();
    for (int off = 128; off > 0; off >>= 1) {
        if (tid < off) {
            if (s_red[tid + off] > s_red[tid] ||
                (s_red[tid + off] == s_red[tid] && s_redi[tid + off] < s_redi[tid])) {
                s_red[tid]  = s_red[tid + off];
                s_redi[tid] = s_redi[tid + off];
            }
        }
        __syncthreads();
    }
    const float lmax = s_red[0];
    const int   argi = s_redi[0];
    __syncthreads();

    // Pass 2: softmax expectations + focal loss vs. Gaussian target.
    const float gx = xy[b * 2 + 0], gy = xy[b * 2 + 1];
    const int cx = iclamp((int)rintf(gx * 0.25f), 0, 127);
    const int cy = iclamp((int)rintf(gy * 0.25f), 0, 127);
    float se = 0.f, sex = 0.f, sey = 0.f, fl = 0.f, np = 0.f;
    for (int i = tid; i < 16384; i += 256) {
        int ix = i & 127, iy = i >> 7;
        float l = bilin(s_log, ix, iy);
        float e = __expf(l - lmax);
        se += e; sex += e * (float)ix; sey += e * (float)iy;
        float p = 1.f / (1.f + __expf(-l));
        p = fminf(fmaxf(p, 1e-6f), 1.f - 1e-6f);
        if (ix == cx && iy == cy) {
            fl += -__logf(p) * (1.f - p) * (1.f - p);
            np += 1.f;
        } else {
            float ddx = (float)(ix - cx), ddy = (float)(iy - cy);
            float tgt = __expf(-0.5f * (ddx * ddx + ddy * ddy) * SIG2I);
            float om = 1.f - tgt, om2 = om * om;
            fl += -__logf(1.f - p) * p * p * om2 * om2;
        }
    }
    auto block_sum = [&](float v) -> float {
        __syncthreads();
        s_red[tid] = v; __syncthreads();
        for (int off = 128; off > 0; off >>= 1) {
            if (tid < off) s_red[tid] += s_red[tid + off];
            __syncthreads();
        }
        return s_red[0];
    };
    float Se = block_sum(se), Sx = block_sum(sex), Sy = block_sum(sey);
    float Fl = block_sum(fl), Np = block_sum(np);

    if (tid == 0) {
        float xs = (Sx / Se) * 4.f, ys = (Sy / Se) * 4.f;
        out[b] = xs; out[64 + b] = ys;
        ws_fl[b] = Fl; ws_np[b] = Np;
        float dxn = fabsf((xs - gx) * (1.f / 511.f));
        float dyn = fabsf((ys - gy) * (1.f / 511.f));
        ws_xy[b] = (dxn < 1.f ? 0.5f * dxn * dxn : dxn - 0.5f)
                 + (dyn < 1.f ? 0.5f * dyn * dyn : dyn - 0.5f);

        int py = argi >> 7, px = argi & 127;
        float wsum = 0.f, wxs = 0.f, wys = 0.f;
        for (int oy = -1; oy <= 1; ++oy)
            for (int ox = -1; ox <= 1; ++ox) {
                int xi = px + ox, yi = py + oy;
                bool valid = (xi >= 0 && xi < 128 && yi >= 0 && yi < 128);
                int xc = iclamp(xi, 0, 127), yc = iclamp(yi, 0, 127);
                float l = bilin(s_log, xc, yc);
                float p = 1.f / (1.f + __expf(-l));
                float wv = valid ? p : 0.f;
                wsum += wv; wxs += wv * (float)xc; wys += wv * (float)yc;
            }
        float ddx = 0.f, ddy = 0.f;
        if (wsum > 1e-6f) {
            float den = fmaxf(wsum, 1e-6f);
            ddx = wxs / den - (float)px;
            ddy = wys / den - (float)py;
        }
        out[128 + b] = ((float)px + ddx) * 4.f;
        out[192 + b] = ((float)py + ddy) * 4.f;
    }
}

// ---------------------------------------------------------------------------
// Kernel 6: combine scalar loss terms.
// ---------------------------------------------------------------------------
__global__ void finalize_kernel(const float* __restrict__ ws_fl,
                                const float* __restrict__ ws_np,
                                const float* __restrict__ ws_xy,
                                const float* __restrict__ ws_z,
                                float* __restrict__ out) {
    __shared__ float s[4][64];
    int t = threadIdx.x;
    s[0][t] = ws_fl[t]; s[1][t] = ws_np[t]; s[2][t] = ws_xy[t]; s[3][t] = ws_z[t];
    __syncthreads();
    if (t == 0) {
        float fl = 0.f, np = 0.f, xyl = 0.f, zl = 0.f;
        for (int i = 0; i < 64; ++i) {
            fl += s[0][i]; np += s[1][i]; xyl += s[2][i]; zl += s[3][i];
        }
        out[320] = fl / fmaxf(np, 1.f) + xyl * (1.f / 128.f) + zl * (1.f / 64.f);
    }
}

// ---------------------------------------------------------------------------
extern "C" void kernel_launch(void* const* d_in, const int* in_sizes, int n_in,
                              void* d_out, int out_size, void* d_ws, size_t ws_size,
                              hipStream_t stream) {
    const float* hm_feats = (const float*)d_in[0];   // (64,112,32,32)
    const float* z_feats  = (const float*)d_in[1];   // (64,960,16,16)
    const float* xy       = (const float*)d_in[2];   // (64,2)
    const float* z        = (const float*)d_in[3];   // (64)
    const float* conv1_w  = (const float*)d_in[4];   // (256,112,3,3)
    const float* conv1_b  = (const float*)d_in[5];   // (256)
    const float* conv2_w  = (const float*)d_in[6];   // (1,256,1,1)
    const float* conv2_b  = (const float*)d_in[7];   // (1)
    const float* fc1_w    = (const float*)d_in[8];   // (256,960)
    const float* fc1_b    = (const float*)d_in[9];   // (256)
    const float* fc2_w    = (const float*)d_in[10];  // (1,256)
    const float* fc2_b    = (const float*)d_in[11];  // (1)
    float* out = (float*)d_out;

    float* ws      = (float*)d_ws;
    float*  logits = ws;                   // 65536 f
    float*  zp     = ws + 65536;           // 61440 f
    float*  h      = ws + 126976;          // 16384 f
    float*  ws_fl  = ws + 143360;          // 64 f
    float*  ws_np  = ws + 143424;          // 64 f
    float*  ws_xy  = ws + 143488;          // 64 f
    float*  ws_z   = ws + 143552;          // 64 f
    __bf16* w1t    = (__bf16*)(ws + 143616);   // 256*1152 bf16 = 576 KB

    wtrans_kernel<<<(256 * KT + 255) / 256, 256, 0, stream>>>(conv1_w, w1t);

    conv_fused_kernel<<<64 * 16, 128, 0, stream>>>(
        hm_feats, w1t, conv1_b, conv2_w, conv2_b, logits);

    zpool_kernel<<<(64 * 960 + 255) / 256, 256, 0, stream>>>(z_feats, zp);

    fc1_kernel<<<64, 32, 0, stream>>>(zp, fc1_w, fc1_b, h);

    fc2_kernel<<<1, 64, 0, stream>>>(h, fc2_w, fc2_b, z, out, ws_z);

    heatmap_kernel<<<64, 256, 0, stream>>>(logits, xy, out, ws_fl, ws_np, ws_xy);

    finalize_kernel<<<1, 64, 0, stream>>>(ws_fl, ws_np, ws_xy, ws_z, out);
}
